// Seq2SeqGenerator_70935679861239
// MI455X (gfx1250) — compile-verified, matrix operand order checked by reading, not run on hardware
//
#include <hip/hip_runtime.h>
#include <hip/hip_bf16.h>
#include <math.h>

// ---------------------------------------------------------------------------
// Seq2Seq (bi-LSTM encoder + attention decoder) for MI455X / gfx1250.
// All GEMMs use v_wmma_f32_16x16x32_bf16 (bf16 A/B, f32 accum).
// Key algebraic hoists (exact):
//   * x@Wih.T lifted out of the recurrent loops (one big GEMM per direction)
//   * attn:  cat([q,enc])@attn_W.T == q@Wq.T + enc@We.T ; enc part precomputed
//   * dec cell0: cat([emb,ctx])@Wih0.T split; emb part precomputed for all t
//   * out_W kept as bf16 (65.5 MB) -> resident in the 192 MB L2 across steps
// GEMM scheme: every M here is a multiple of 32, so each wave owns a 32x16 C
// strip (two 16x16 accumulators) and loads each weight fragment ONCE per two
// WMMAs -> halves weight-side L2 traffic on the dominant out-projection.
// ---------------------------------------------------------------------------

typedef __bf16 bf16;
typedef __attribute__((ext_vector_type(16))) __bf16 v16bf;
typedef __attribute__((ext_vector_type(8)))  float  v8f;

union FragBF { v16bf v; uint4 q[2]; };

// Load one 16x32 bf16 WMMA operand tile (A: rows=M, or B from row-major [N,K]
// weight so that B[k][n] = W[n][k]).  CDNA5 layout: lanes 0-15 hold row, sel
// bit (lane>=16) selects K half-groups {0..7,16..23} vs {8..15,24..31} ->
// exactly two contiguous 16-byte loads per lane.
__device__ __forceinline__ v16bf load_frag_bf16(const bf16* __restrict__ base,
                                                int rowBase, int ld, int kBase,
                                                int lane) {
  int r   = rowBase + (lane & 15);
  int sel = (lane >> 4) & 1;
  const bf16* p = base + (size_t)r * ld + kBase + sel * 8;
  FragBF f;
  f.q[0] = *(const uint4*)(p);        // K = sel*8 .. sel*8+7
  f.q[1] = *(const uint4*)(p + 16);   // K = 16+sel*8 .. 16+sel*8+7
  return f.v;
}

// C[M,N] = A[M,K] @ W[N,K]^T (+ add0[M,N]) (+ bias0[N]) (+ bias1[N])
// One 32x16 C strip per wave: the B fragment is loaded once and fed to two
// WMMAs (M-tile m0 and m0+16).  Requires M % 32 == 0, N % 16, K % 32.
__global__ void gemm_bf16_wmma(const bf16* __restrict__ A, int lda,
                               const bf16* __restrict__ W, int ldw,
                               float* __restrict__ C, int ldc,
                               const float* __restrict__ add0,
                               const float* __restrict__ bias0,
                               const float* __restrict__ bias1,
                               int M, int N, int K) {
  const int wavesPerBlock = blockDim.x >> 5;
  const int wave = threadIdx.x >> 5;
  const int lane = threadIdx.x & 31;
  const int mPairs = M >> 5, nTiles = N >> 4;
  const int tile = blockIdx.x * wavesPerBlock + wave;
  if (tile >= mPairs * nTiles) return;          // wave-uniform: EXEC all-ones
  const int mp = tile / nTiles, nt = tile % nTiles;
  const int m0 = mp << 5,       n0 = nt << 4;

  v8f acc0 = {0.f, 0.f, 0.f, 0.f, 0.f, 0.f, 0.f, 0.f};
  v8f acc1 = {0.f, 0.f, 0.f, 0.f, 0.f, 0.f, 0.f, 0.f};
  for (int k0 = 0; k0 < K; k0 += 32) {
    v16bf fb  = load_frag_bf16(W, n0, ldw, k0, lane);      // shared weight frag
    v16bf fa0 = load_frag_bf16(A, m0,      lda, k0, lane);
    v16bf fa1 = load_frag_bf16(A, m0 + 16, lda, k0, lane);
    acc0 = __builtin_amdgcn_wmma_f32_16x16x32_bf16(
        false, fa0, false, fb, (short)0, acc0, false, false);
    acc1 = __builtin_amdgcn_wmma_f32_16x16x32_bf16(
        false, fa1, false, fb, (short)0, acc1, false, false);
  }

  const int n   = n0 + (lane & 15);
  const int sel = (lane >> 4) & 1;
  float badd = 0.f;
  if (bias0) badd += bias0[n];
  if (bias1) badd += bias1[n];
#pragma unroll
  for (int r = 0; r < 8; ++r) {
    const int m = m0 + r + (sel << 3);          // C layout: VGPR r -> M=r(+8)
    float v0 = acc0[r] + badd;
    float v1 = acc1[r] + badd;
    if (add0) {
      v0 += add0[(size_t)m * ldc + n];
      v1 += add0[(size_t)(m + 16) * ldc + n];
    }
    C[(size_t)m * ldc + n]        = v0;
    C[(size_t)(m + 16) * ldc + n] = v1;
  }
}

// z[B,4H] -> (h2,c2); PyTorch gate order i,f,g,o.  Optionally stores / adds
// h2 into an extra f32 destination (encoder outputs), always mirrors h2->bf16.
__global__ void lstm_gates_kernel(const float* __restrict__ z,
                                  float* __restrict__ h_state,
                                  float* __restrict__ c_state,
                                  bf16*  __restrict__ h_bf,
                                  float* __restrict__ h_out, int accumulate,
                                  int B, int H) {
  int i = blockIdx.x * blockDim.x + threadIdx.x;
  if (i >= B * H) return;
  int b = i / H, hh = i - b * H;
  const float* zr = z + (size_t)b * 4 * H;
  float zi = zr[hh], zf = zr[H + hh], zg = zr[2 * H + hh], zo = zr[3 * H + hh];
  float si = 1.f / (1.f + __expf(-zi));
  float sf = 1.f / (1.f + __expf(-zf));
  float so = 1.f / (1.f + __expf(-zo));
  float c2 = sf * c_state[i] + si * tanhf(zg);
  float h2 = so * tanhf(c2);
  c_state[i] = c2;
  h_state[i] = h2;
  h_bf[i] = (bf16)h2;
  if (h_out) { if (accumulate) h_out[i] += h2; else h_out[i] = h2; }
}

// Per-batch attention: scores -> softmax -> context. One workgroup per b.
__global__ void attn_kernel(const float* __restrict__ qproj,    // [B,H]
                            const float* __restrict__ encProj,  // [S*B,H] (+b)
                            const float* __restrict__ encOut,   // [S*B,H]
                            const float* __restrict__ attn_v,   // [H]
                            float* __restrict__ ctx,            // [B,H]
                            bf16*  __restrict__ ctx_bf,         // [B,H]
                            int S, int B, int H) {
  __shared__ float red[256];
  __shared__ float sc[128];
  const int b = blockIdx.x, tid = threadIdx.x;
  for (int s = 0; s < S; ++s) {
    float p = 0.f;
    for (int h = tid; h < H; h += blockDim.x) {
      float e = tanhf(qproj[b * H + h] + encProj[(size_t)(s * B + b) * H + h]);
      p += e * attn_v[h];
    }
    red[tid] = p; __syncthreads();
    for (int off = 128; off > 0; off >>= 1) {
      if (tid < off) red[tid] += red[tid + off];
      __syncthreads();
    }
    if (tid == 0) sc[s] = red[0];
    __syncthreads();
  }
  float mx = -3.4e38f;
  for (int s = tid; s < S; s += blockDim.x) mx = fmaxf(mx, sc[s]);
  red[tid] = mx; __syncthreads();
  for (int off = 128; off > 0; off >>= 1) {
    if (tid < off) red[tid] = fmaxf(red[tid], red[tid + off]);
    __syncthreads();
  }
  mx = red[0]; __syncthreads();
  float sm = 0.f;
  for (int s = tid; s < S; s += blockDim.x) sm += __expf(sc[s] - mx);
  red[tid] = sm; __syncthreads();
  for (int off = 128; off > 0; off >>= 1) {
    if (tid < off) red[tid] += red[tid + off];
    __syncthreads();
  }
  sm = red[0]; __syncthreads();
  const float inv = 1.f / sm;
  for (int s = tid; s < S; s += blockDim.x) sc[s] = __expf(sc[s] - mx) * inv;
  __syncthreads();
  for (int h = tid; h < H; h += blockDim.x) {
    float a = 0.f;
    for (int s = 0; s < S; ++s) a += sc[s] * encOut[(size_t)(s * B + b) * H + h];
    ctx[b * H + h] = a;
    ctx_bf[b * H + h] = (bf16)a;
  }
}

// log_softmax over V per row; one workgroup per b; writes directly to d_out.
__global__ void log_softmax_kernel(const float* __restrict__ logits,
                                   float* __restrict__ out, int Vsz) {
  __shared__ float red[256];
  const int b = blockIdx.x, tid = threadIdx.x;
  const float* row = logits + (size_t)b * Vsz;
  float mx = -3.4e38f;
  for (int v = tid; v < Vsz; v += blockDim.x) mx = fmaxf(mx, row[v]);
  red[tid] = mx; __syncthreads();
  for (int off = 128; off > 0; off >>= 1) {
    if (tid < off) red[tid] = fmaxf(red[tid], red[tid + off]);
    __syncthreads();
  }
  mx = red[0]; __syncthreads();
  float sm = 0.f;
  for (int v = tid; v < Vsz; v += blockDim.x) sm += __expf(row[v] - mx);
  red[tid] = sm; __syncthreads();
  for (int off = 128; off > 0; off >>= 1) {
    if (tid < off) red[tid] += red[tid + off];
    __syncthreads();
  }
  const float ls = mx + __logf(red[0]);
  float* orow = out + (size_t)b * Vsz;
  for (int v = tid; v < Vsz; v += blockDim.x) orow[v] = row[v] - ls;
}

// f32 -> bf16 with row slicing (extract column block [soff, soff+cols) )
__global__ void conv_bf16_slice(const float* __restrict__ src,
                                bf16* __restrict__ dst,
                                int rows, int cols, int src_ld, int src_off) {
  int i = blockIdx.x * blockDim.x + threadIdx.x;
  if (i >= rows * cols) return;
  int r = i / cols, c = i - r * cols;
  dst[i] = (bf16)src[(size_t)r * src_ld + src_off + c];
}

__global__ void embed_gather_bf16(const int* __restrict__ toks,
                                  const float* __restrict__ embed,
                                  bf16* __restrict__ out, int n, int E) {
  int i = blockIdx.x * blockDim.x + threadIdx.x;
  if (i >= n * E) return;
  int t = i / E, e = i - t * E;
  out[i] = (bf16)embed[(size_t)toks[t] * E + e];
}

__global__ void zero_f32_kernel(float* __restrict__ p, size_t n) {
  size_t i = (size_t)blockIdx.x * blockDim.x + threadIdx.x;
  if (i < n) p[i] = 0.f;
}

// out[b, 0:H] = a[b,:] ; out[b, H:2H] = c[b,:]
__global__ void concat_bf16(const bf16* __restrict__ a,
                            const bf16* __restrict__ c,
                            bf16* __restrict__ out, int B, int H) {
  int i = blockIdx.x * blockDim.x + threadIdx.x;
  if (i >= B * 2 * H) return;
  int b = i / (2 * H), h = i - b * 2 * H;
  out[i] = (h < H) ? a[b * H + h] : c[b * H + (h - H)];
}

extern "C" void kernel_launch(void* const* d_in, const int* in_sizes, int n_in,
                              void* d_out, int out_size, void* d_ws, size_t ws_size,
                              hipStream_t stream) {
  (void)in_sizes; (void)n_in; (void)out_size; (void)ws_size;
  constexpr int V = 32000, E = 256, H = 512, S = 128, B = 32, T = 32;
  constexpr int G = 4 * H;      // 2048 gate width

  const int*   src      = (const int*)  d_in[0];
  const int*   trg      = (const int*)  d_in[1];
  const float* embed    = (const float*)d_in[2];
  const float* eWih_f   = (const float*)d_in[3];
  const float* eWhh_f   = (const float*)d_in[4];
  const float* ebih_f   = (const float*)d_in[5];
  const float* ebhh_f   = (const float*)d_in[6];
  const float* eWih_b   = (const float*)d_in[7];
  const float* eWhh_b   = (const float*)d_in[8];
  const float* ebih_b   = (const float*)d_in[9];
  const float* ebhh_b   = (const float*)d_in[10];
  const float* attn_W   = (const float*)d_in[11];
  const float* attn_b   = (const float*)d_in[12];
  const float* attn_v   = (const float*)d_in[13];
  const float* dWih0    = (const float*)d_in[14];
  const float* dWhh0    = (const float*)d_in[15];
  const float* dbih0    = (const float*)d_in[16];
  const float* dbhh0    = (const float*)d_in[17];
  const float* dWih1    = (const float*)d_in[18];
  const float* dWhh1    = (const float*)d_in[19];
  const float* dbih1    = (const float*)d_in[20];
  const float* dbhh1    = (const float*)d_in[21];
  const float* out_W    = (const float*)d_in[22];
  const float* out_b    = (const float*)d_in[23];
  float* out = (float*)d_out;

  // ---- workspace carve-up (256B aligned) --------------------------------
  char* wsp = (char*)d_ws; size_t off = 0;
  auto alloc = [&](size_t bytes) -> void* {
    void* p = wsp + off; off = (off + bytes + 255) & ~(size_t)255; return p;
  };
  bf16* WihF   = (bf16*)alloc((size_t)G * E * 2);
  bf16* WihB   = (bf16*)alloc((size_t)G * E * 2);
  bf16* WhhF   = (bf16*)alloc((size_t)G * H * 2);
  bf16* WhhB   = (bf16*)alloc((size_t)G * H * 2);
  bf16* Wq     = (bf16*)alloc((size_t)H * H * 2);
  bf16* We     = (bf16*)alloc((size_t)H * H * 2);
  bf16* Wih0E  = (bf16*)alloc((size_t)G * E * 2);
  bf16* Wih0C  = (bf16*)alloc((size_t)G * H * 2);
  bf16* Whh0b  = (bf16*)alloc((size_t)G * H * 2);
  bf16* Wih1b  = (bf16*)alloc((size_t)G * H * 2);
  bf16* Whh1b  = (bf16*)alloc((size_t)G * H * 2);
  bf16* OutWb  = (bf16*)alloc((size_t)V * 2 * H * 2);       // 65.5 MB, L2-resident
  bf16* embSrc = (bf16*)alloc((size_t)S * B * E * 2);
  bf16* embTrg = (bf16*)alloc((size_t)(T - 1) * B * E * 2);
  float* xWf     = (float*)alloc((size_t)S * B * G * 4);
  float* xWb     = (float*)alloc((size_t)S * B * G * 4);
  float* decEmbW = (float*)alloc((size_t)(T - 1) * B * G * 4);
  float* encOut  = (float*)alloc((size_t)S * B * H * 4);
  bf16*  encOutB = (bf16*) alloc((size_t)S * B * H * 2);
  float* encProj = (float*)alloc((size_t)S * B * H * 4);
  float* zbuf    = (float*)alloc((size_t)B * G * 4);
  float* hF      = (float*)alloc((size_t)B * H * 4);
  float* cF      = (float*)alloc((size_t)B * H * 4);
  bf16*  hFb     = (bf16*) alloc((size_t)B * H * 2);
  float* hB      = (float*)alloc((size_t)B * H * 4);
  float* cB      = (float*)alloc((size_t)B * H * 4);
  bf16*  hBb     = (bf16*) alloc((size_t)B * H * 2);
  float* qproj   = (float*)alloc((size_t)B * H * 4);
  float* ctx     = (float*)alloc((size_t)B * H * 4);
  bf16*  ctxB    = (bf16*) alloc((size_t)B * H * 2);
  bf16*  catBuf  = (bf16*) alloc((size_t)B * 2 * H * 2);
  float* logits  = (float*)alloc((size_t)B * V * 4);

  auto gemm = [&](const bf16* A, int lda, const bf16* Wm, int ldw,
                  float* C, int ldc, const float* add0,
                  const float* b0, const float* b1, int M, int N, int K) {
    int tiles = (M >> 5) * (N >> 4);            // 32x16 strips
    gemm_bf16_wmma<<<(tiles + 3) / 4, 128, 0, stream>>>(
        A, lda, Wm, ldw, C, ldc, add0, b0, b1, M, N, K);
  };
  auto conv = [&](const float* s, bf16* d, int rows, int cols, int sld, int soff) {
    int n = rows * cols;
    conv_bf16_slice<<<(n + 255) / 256, 256, 0, stream>>>(s, d, rows, cols, sld, soff);
  };
  auto zero = [&](void* p, size_t nfloats) {
    zero_f32_kernel<<<(int)((nfloats + 255) / 256), 256, 0, stream>>>((float*)p, nfloats);
  };
  auto gates = [&](float* h, float* c, bf16* hb, float* hout, int acc) {
    lstm_gates_kernel<<<(B * H + 255) / 256, 256, 0, stream>>>(
        zbuf, h, c, hb, hout, acc, B, H);
  };

  // ---- weight conversion / slicing --------------------------------------
  conv(eWih_f, WihF, G, E, E, 0);
  conv(eWih_b, WihB, G, E, E, 0);
  conv(eWhh_f, WhhF, G, H, H, 0);
  conv(eWhh_b, WhhB, G, H, H, 0);
  conv(attn_W, Wq, H, H, 2 * H, 0);
  conv(attn_W, We, H, H, 2 * H, H);
  conv(dWih0, Wih0E, G, E, H + E, 0);
  conv(dWih0, Wih0C, G, H, H + E, E);
  conv(dWhh0, Whh0b, G, H, H, 0);
  conv(dWih1, Wih1b, G, H, H, 0);
  conv(dWhh1, Whh1b, G, H, H, 0);
  conv(out_W, OutWb, V, 2 * H, 2 * H, 0);

  // ---- embeddings + hoisted input projections ---------------------------
  embed_gather_bf16<<<(S * B * E + 255) / 256, 256, 0, stream>>>(src, embed, embSrc, S * B, E);
  embed_gather_bf16<<<((T - 1) * B * E + 255) / 256, 256, 0, stream>>>(trg, embed, embTrg, (T - 1) * B, E);
  gemm(embSrc, E, WihF, E, xWf, G, nullptr, nullptr, nullptr, S * B, G, E);
  gemm(embSrc, E, WihB, E, xWb, G, nullptr, nullptr, nullptr, S * B, G, E);
  gemm(embTrg, E, Wih0E, E, decEmbW, G, nullptr, nullptr, nullptr, (T - 1) * B, G, E);

  // ---- zero initial LSTM state ------------------------------------------
  zero(hFb, (size_t)B * H / 2); zero(cF, (size_t)B * H);
  zero(hBb, (size_t)B * H / 2); zero(cB, (size_t)B * H);
  zero(out, (size_t)B * V);                     // outputs[0] = 0

  // ---- encoder: forward then backward (serial recurrences) --------------
  for (int t = 0; t < S; ++t) {
    gemm(hFb, H, WhhF, H, zbuf, G, xWf + (size_t)t * B * G, ebih_f, ebhh_f, B, G, H);
    gates(hF, cF, hFb, encOut + (size_t)t * B * H, 0);
  }
  for (int k = 0; k < S; ++k) {
    int p = S - 1 - k;                          // b_outs[::-1] fold-in
    gemm(hBb, H, WhhB, H, zbuf, G, xWb + (size_t)p * B * G, ebih_b, ebhh_b, B, G, H);
    gates(hB, cB, hBb, encOut + (size_t)p * B * H, 1);
  }

  // ---- attention precompute: encProj = encOut @ We.T + attn_b -----------
  conv(encOut, encOutB, S * B, H, H, 0);
  gemm(encOutB, H, We, H, encProj, H, nullptr, attn_b, nullptr, S * B, H, H);

  // ---- decoder: layer0 state := (hF,cF), layer1 state := (hB,cB) --------
  for (int t = 0; t < T - 1; ++t) {
    // q = h1 ; qproj = q @ Wq.T
    gemm(hBb, H, Wq, H, qproj, H, nullptr, nullptr, nullptr, B, H, H);
    attn_kernel<<<B, 256, 0, stream>>>(qproj, encProj, encOut, attn_v, ctx, ctxB, S, B, H);
    // cell0: z = decEmbW[t] + ctx@Wih0C.T + h0@Whh0.T + bih0 + bhh0
    gemm(ctxB, H, Wih0C, H, zbuf, G, decEmbW + (size_t)t * B * G, dbih0, dbhh0, B, G, H);
    gemm(hFb, H, Whh0b, H, zbuf, G, zbuf, nullptr, nullptr, B, G, H);
    gates(hF, cF, hFb, nullptr, 0);
    // cell1: z = h0'@Wih1.T + h1@Whh1.T + bih1 + bhh1
    gemm(hFb, H, Wih1b, H, zbuf, G, nullptr, dbih1, dbhh1, B, G, H);
    gemm(hBb, H, Whh1b, H, zbuf, G, zbuf, nullptr, nullptr, B, G, H);
    gates(hB, cB, hBb, nullptr, 0);
    // logits = [h1'|ctx] @ out_W.T + out_b ; then log_softmax -> out[t+1]
    concat_bf16<<<(B * 2 * H + 255) / 256, 256, 0, stream>>>(hBb, ctxB, catBuf, B, H);
    gemm(catBuf, 2 * H, OutWb, 2 * H, logits, V, nullptr, out_b, nullptr, B, V, 2 * H);
    log_softmax_kernel<<<B, 256, 0, stream>>>(logits, out + (size_t)(t + 1) * B * V, V);
  }
}
